// STGNOT_ANALYSIS_21071109554438
// MI455X (gfx1250) — matrix operand order repaired and checked by reading, not compiled, not run on hardware
//
#include <hip/hip_runtime.h>
#include <hip/hip_bf16.h>

// ---------------------------------------------------------------------------
// Types
// ---------------------------------------------------------------------------
typedef __bf16 bf16;
typedef __attribute__((ext_vector_type(16))) __bf16 v16bf;
typedef __attribute__((ext_vector_type(8)))  __bf16 v8bf;
typedef __attribute__((ext_vector_type(8)))  float  v8f;
typedef __attribute__((ext_vector_type(4)))  int    v4i;

// ---------------------------------------------------------------------------
// Problem constants (from the reference)
// ---------------------------------------------------------------------------
#define Lc   3
#define Ec   4
#define NINc 2
#define Cc   256
#define Hc   8
#define Ic   1024
#define SDc  2
#define OUTc 3
#define MLc  2
#define Bc   2
#define T1c  2048
#define T2c  2048
#define HDc  32
#define MR   4096          // B*T1 == B*T2 token rows
#define LDST 40            // padded LDS row stride (elements): 80B, 16B-aligned

// ---------------------------------------------------------------------------
// CDNA5 async-copy (confirmed working on this toolchain in round 2)
// ---------------------------------------------------------------------------
#if __has_builtin(__builtin_amdgcn_global_load_async_to_lds_b128)
#define HAVE_ASYNC 1
typedef __attribute__((address_space(1))) v4i* gas_v4i;
typedef __attribute__((address_space(3))) v4i* las_v4i;
__device__ __forceinline__ void async_b128(const bf16* g, bf16* l) {
  __builtin_amdgcn_global_load_async_to_lds_b128(
      (gas_v4i)const_cast<bf16*>(g), (las_v4i)l, 0, 0);
}
__device__ __forceinline__ void wait_async0() {
#if __has_builtin(__builtin_amdgcn_s_wait_asynccnt)
  __builtin_amdgcn_s_wait_asynccnt(0);
#else
  asm volatile("s_wait_asynccnt 0x0" ::: "memory");
#endif
}
#endif

__device__ __forceinline__ float gelu_f(float x) {
  return 0.5f * x * (1.f + erff(x * 0.70710678118654752f));
}

__device__ __forceinline__ v16bf cat8(v8bf lo, v8bf hi) {
  return __builtin_shufflevector(lo, hi, 0,1,2,3,4,5,6,7,8,9,10,11,12,13,14,15);
}

// ---------------------------------------------------------------------------
// WMMA GEMM: C[M,N] = epilogue(A[M,K] @ Bt^T[K,N] + bias [+ res])
// A bf16 (M,K) row-major; Bt bf16 (N,K) row-major (pre-transposed B).
// Double-buffered LDS, async global->LDS staging, bf16 WMMA, f32 epilogue.
// Block 256 threads = 8 waves (wave32); block tile 128x64; wave tile 32x32.
// ---------------------------------------------------------------------------
__global__ void __launch_bounds__(256)
k_gemm(const bf16* __restrict__ A, const bf16* __restrict__ Bt,
       const float* __restrict__ bias, float* __restrict__ C,
       const float* __restrict__ res, int M, int K, int N, int flags)
{
  __shared__ __align__(16) bf16 la[2][128 * LDST];
  __shared__ __align__(16) bf16 lb[2][64 * LDST];
  const int tid  = threadIdx.x;
  const int lane = tid & 31;
  const int wid  = tid >> 5;
  const int wr   = wid & 3;            // wave row (4)
  const int wc   = wid >> 2;           // wave col (2)
  const int r16  = lane & 15;
  const int lg   = lane >> 4;
  const int m0   = blockIdx.y * 128;
  const int n0   = blockIdx.x * 64;

  const v8f vz = {0.f,0.f,0.f,0.f,0.f,0.f,0.f,0.f};
  v8f acc[2][2];
  acc[0][0] = vz; acc[0][1] = vz; acc[1][0] = vz; acc[1][1] = vz;

  const int arow = tid >> 1;           // A: 128 rows, 2 threads/row, 16 elems each
  const int aseg = (tid & 1) * 16;
  const int brow = tid >> 2;           // B: 64 rows, 4 threads/row, 8 elems each
  const int bseg = (tid & 3) * 8;

  auto stage = [&](int kk0, int bsel) {
    bf16* lA = &la[bsel][0];
    bf16* lB = &lb[bsel][0];
#ifdef HAVE_ASYNC
    const bool aok = (m0 + 128 <= M) && (kk0 + 32 <= K) && ((K & 7) == 0);
    if (aok) {
      const bf16* gp = A + (size_t)(m0 + arow) * K + kk0 + aseg;
      async_b128(gp,     &lA[arow * LDST + aseg]);
      async_b128(gp + 8, &lA[arow * LDST + aseg + 8]);
    } else
#endif
    {
      #pragma unroll 1
      for (int j = 0; j < 16; ++j) {
        int kk = kk0 + aseg + j;
        bf16 v = (bf16)0.f;
        if ((m0 + arow) < M && kk < K) v = A[(size_t)(m0 + arow) * K + kk];
        lA[arow * LDST + aseg + j] = v;
      }
    }
#ifdef HAVE_ASYNC
    const bool bok = (n0 + 64 <= N) && (kk0 + 32 <= K) && ((K & 7) == 0);
    if (bok) {
      const bf16* gp = Bt + (size_t)(n0 + brow) * K + kk0 + bseg;
      async_b128(gp, &lB[brow * LDST + bseg]);
    } else
#endif
    {
      #pragma unroll 1
      for (int j = 0; j < 8; ++j) {
        int kk = kk0 + bseg + j;
        bf16 v = (bf16)0.f;
        if ((n0 + brow) < N && kk < K) v = Bt[(size_t)(n0 + brow) * K + kk];
        lB[brow * LDST + bseg + j] = v;
      }
    }
  };

  stage(0, 0);
#ifdef HAVE_ASYNC
  wait_async0();
#endif
  __syncthreads();

  int buf = 0;
  for (int k0 = 0; k0 < K; k0 += 32) {
    if (k0 + 32 < K) stage(k0 + 32, buf ^ 1);   // prefetch next tile (async)

    // ---- fragments per ISA 7.12.2 bf16 layouts ----
    const bf16* lA = &la[buf][0];
    const bf16* lB = &lb[buf][0];
    v16bf af[2], bfrag[2];
    #pragma unroll
    for (int t = 0; t < 2; ++t) {
      int r  = wr * 32 + t * 16 + r16;
      int kc = lg * 8;                 // A: K {0..7,16..23} / {8..15,24..31}
      v8bf alo = *(const v8bf*)&lA[r * LDST + kc];
      v8bf ahi = *(const v8bf*)&lA[r * LDST + kc + 16];
      af[t] = cat8(alo, ahi);
      int n  = wc * 32 + t * 16 + r16;
      int kb = lg * 8;                 // B: K {0..7,16..23} / {8..15,24..31}
      v8bf blo = *(const v8bf*)&lB[n * LDST + kb];
      v8bf bhi = *(const v8bf*)&lB[n * LDST + kb + 16];
      bfrag[t] = cat8(blo, bhi);
    }
    #pragma unroll
    for (int tm = 0; tm < 2; ++tm)
      #pragma unroll
      for (int tn = 0; tn < 2; ++tn)
        acc[tm][tn] = __builtin_amdgcn_wmma_f32_16x16x32_bf16(
            false, af[tm], false, bfrag[tn], (short)0, acc[tm][tn], false, false);

#ifdef HAVE_ASYNC
    wait_async0();
#endif
    __syncthreads();
    buf ^= 1;
  }

  // ---- epilogue (C layout: row = v + 8*lg, col = r16 per 16x16 tile) ----
  #pragma unroll
  for (int tm = 0; tm < 2; ++tm) {
    #pragma unroll
    for (int tn = 0; tn < 2; ++tn) {
      int col = n0 + wc * 32 + tn * 16 + r16;
      if (col >= N) continue;
      #pragma unroll
      for (int v = 0; v < 8; ++v) {
        int row = m0 + wr * 32 + tm * 16 + v + lg * 8;
        if (row >= M) continue;
        float xv = acc[tm][tn][v];
        if (bias) xv += bias[col];
        if (flags & 1) xv = gelu_f(xv);
        if (res) xv += res[(size_t)row * N + col];
        C[(size_t)row * N + col] = xv;
      }
    }
  }
}

// ---------------------------------------------------------------------------
// Flash attention, HD=32.  Q,K bf16 (B,T,H*HD); V pre-transposed bf16
// (B,H,HD,T); O f32 accumulated.  Grid: (Tq/128, B*H).  8 waves/block,
// 16 queries/wave.  32 keys/chunk: S = 2 WMMAs (reduce = HD = 32), online
// softmax, P via wave-local LDS layout flip, PV = 2 WMMAs.  K/V tiles are
// double-buffered; waves 0-3 async-stage K, waves 4-7 async-stage V.
// ---------------------------------------------------------------------------
__global__ void __launch_bounds__(256)
k_attn(const bf16* __restrict__ Qb, const bf16* __restrict__ Kb,
       const bf16* __restrict__ Vt, float* __restrict__ O,
       int Tq, int Tk, int Cd, int nheads, float scale)
{
  __shared__ __align__(16) bf16 lk[2][32 * LDST];   // [key][hd]
  __shared__ __align__(16) bf16 lv[2][32 * LDST];   // [hd][key]
  __shared__ float lp[8 * 16 * LDST];               // per-wave P scratch
  const int tid  = threadIdx.x;
  const int lane = tid & 31;
  const int wid  = tid >> 5;
  const int r16  = lane & 15;
  const int lg   = lane >> 4;
  const int b    = blockIdx.y / nheads;
  const int h    = blockIdx.y % nheads;
  const int q0   = blockIdx.x * 128 + wid * 16;
  const size_t qrow = (size_t)b * Tq + q0;

  // Q fragment (A layout), loaded once
  v16bf qf;
  {
    const int kc = lg * 8;
    const size_t base = (qrow + r16) * Cd + h * HDc;
    v8bf qlo = *(const v8bf*)(Qb + base + kc);
    v8bf qhi = *(const v8bf*)(Qb + base + kc + 16);
    qf = cat8(qlo, qhi);
  }

  const v8f vz = {0.f,0.f,0.f,0.f,0.f,0.f,0.f,0.f};
  v8f oacc[2]; oacc[0] = vz; oacc[1] = vz;
  float rmax[8], rsum[8];
  #pragma unroll
  for (int v = 0; v < 8; ++v) { rmax[v] = -1e30f; rsum[v] = 0.f; }

  float* lpw = &lp[wid * 16 * LDST];

  // waves 0-3 stage the K tile, waves 4-7 the V tile (1 b128 per thread)
  auto stage = [&](int cc0, int bsel) {
    if (tid < 128) {
      int key = tid >> 2, seg = (tid & 3) * 8;
      const bf16* gp = Kb + ((size_t)b * Tk + cc0 + key) * Cd + h * HDc + seg;
      bf16* lp_ = &lk[bsel][key * LDST + seg];
#ifdef HAVE_ASYNC
      async_b128(gp, lp_);
#else
      *(v8bf*)lp_ = *(const v8bf*)gp;
#endif
    } else {
      int t2 = tid - 128;
      int hd = t2 >> 2, seg = (t2 & 3) * 8;
      const bf16* gp = Vt + ((size_t)(b * nheads + h) * HDc + hd) * Tk + cc0 + seg;
      bf16* lp_ = &lv[bsel][hd * LDST + seg];
#ifdef HAVE_ASYNC
      async_b128(gp, lp_);
#else
      *(v8bf*)lp_ = *(const v8bf*)gp;
#endif
    }
  };

  stage(0, 0);
#ifdef HAVE_ASYNC
  wait_async0();
#endif
  __syncthreads();

  int buf = 0;
  for (int c0 = 0; c0 < Tk; c0 += 32) {
    if (c0 + 32 < Tk) stage(c0 + 32, buf ^ 1);    // prefetch next K/V tiles

    const bf16* lK = &lk[buf][0];
    const bf16* lV = &lv[buf][0];

    // S = Q Kt : two 16x16 score tiles
    v8f s[2];
    #pragma unroll
    for (int t = 0; t < 2; ++t) {
      int key = t * 16 + r16;
      int kb  = lg * 8;
      v8bf klo = *(const v8bf*)&lK[key * LDST + kb];
      v8bf khi = *(const v8bf*)&lK[key * LDST + kb + 16];
      v16bf kf = cat8(klo, khi);
      s[t] = __builtin_amdgcn_wmma_f32_16x16x32_bf16(
          false, qf, false, kf, (short)0, vz, false, false);
    }

    // online softmax per row (row r = v + 8*lg, 16 lanes per row-group)
    #pragma unroll
    for (int v = 0; v < 8; ++v) {
      float s0 = s[0][v] * scale;
      float s1 = s[1][v] * scale;
      float mv = fmaxf(s0, s1);
      mv = fmaxf(mv, __shfl_xor(mv, 1, 32));
      mv = fmaxf(mv, __shfl_xor(mv, 2, 32));
      mv = fmaxf(mv, __shfl_xor(mv, 4, 32));
      mv = fmaxf(mv, __shfl_xor(mv, 8, 32));
      float nm   = fmaxf(rmax[v], mv);
      float corr = __expf(rmax[v] - nm);
      float p0 = __expf(s0 - nm);
      float p1 = __expf(s1 - nm);
      float ps = p0 + p1;
      ps += __shfl_xor(ps, 1, 32);
      ps += __shfl_xor(ps, 2, 32);
      ps += __shfl_xor(ps, 4, 32);
      ps += __shfl_xor(ps, 8, 32);
      rsum[v] = rsum[v] * corr + ps;
      rmax[v] = nm;
      oacc[0][v] *= corr;
      oacc[1][v] *= corr;
      int prow = v + lg * 8;
      lpw[prow * LDST + r16]      = p0;
      lpw[prow * LDST + 16 + r16] = p1;
    }
    // wave-local LDS RAW: lanes are lockstep, just drain DScnt
    asm volatile("s_wait_dscnt 0x0" ::: "memory");

    // reload P in A layout as bf16
    v16bf pf;
    {
      const int kc2 = lg * 8;
      const float* pr = &lpw[r16 * LDST];
      #pragma unroll
      for (int j = 0; j < 8; ++j) {
        pf[j]     = (bf16)pr[kc2 + j];
        pf[8 + j] = (bf16)pr[kc2 + 16 + j];
      }
    }
    // O += P V
    #pragma unroll
    for (int tn = 0; tn < 2; ++tn) {
      int hd = tn * 16 + r16;
      int kb = lg * 8;
      v8bf vlo = *(const v8bf*)&lV[hd * LDST + kb];
      v8bf vhi = *(const v8bf*)&lV[hd * LDST + kb + 16];
      v16bf vf = cat8(vlo, vhi);
      oacc[tn] = __builtin_amdgcn_wmma_f32_16x16x32_bf16(
          false, pf, false, vf, (short)0, oacc[tn], false, false);
    }
#ifdef HAVE_ASYNC
    wait_async0();
#endif
    __syncthreads();
    buf ^= 1;
  }

  #pragma unroll
  for (int tn = 0; tn < 2; ++tn) {
    int col = h * HDc + tn * 16 + r16;
    #pragma unroll
    for (int v = 0; v < 8; ++v) {
      int prow = v + lg * 8;
      O[(qrow + prow) * Cd + col] += oacc[tn][v] / rsum[v];
    }
  }
}

// ---------------------------------------------------------------------------
// LayerNorm over C=256, one wave per row.  Y = ln(X)*G + Bt (+ res)
// ---------------------------------------------------------------------------
__global__ void __launch_bounds__(256)
k_layernorm(const float* __restrict__ X, const float* __restrict__ G,
            const float* __restrict__ Bt, const float* __restrict__ res,
            float* __restrict__ Y, int rows, int Cd)
{
  int wid  = threadIdx.x >> 5;
  int lane = threadIdx.x & 31;
  int row  = blockIdx.x * 8 + wid;
  if (row >= rows) return;
  const float* xr = X + (size_t)row * Cd;
  float vals[8];
  float sum = 0.f;
  #pragma unroll
  for (int j = 0; j < 8; ++j) { vals[j] = xr[lane + j * 32]; sum += vals[j]; }
  #pragma unroll
  for (int o = 16; o > 0; o >>= 1) sum += __shfl_xor(sum, o, 32);
  float mean = sum * (1.f / 256.f);
  float var = 0.f;
  #pragma unroll
  for (int j = 0; j < 8; ++j) { float d = vals[j] - mean; var += d * d; }
  #pragma unroll
  for (int o = 16; o > 0; o >>= 1) var += __shfl_xor(var, o, 32);
  var *= (1.f / 256.f);
  float rstd = rsqrtf(var + 1e-5f);
  #pragma unroll
  for (int j = 0; j < 8; ++j) {
    int c = lane + j * 32;
    float y = (vals[j] - mean) * rstd * G[c] + Bt[c];
    if (res) y += res[(size_t)row * Cd + c];
    Y[(size_t)row * Cd + c] = y;
  }
}

// ---------------------------------------------------------------------------
// Elementwise helpers
// ---------------------------------------------------------------------------
__global__ void k_cast(const float* __restrict__ s, bf16* __restrict__ d, int n) {
  int i = blockIdx.x * blockDim.x + threadIdx.x;
  if (i < n) d[i] = (bf16)s[i];
}
// (K,N) f32 -> (N,K) bf16
__global__ void k_castT(const float* __restrict__ s, bf16* __restrict__ d, int K, int N) {
  int i = blockIdx.x * blockDim.x + threadIdx.x;
  if (i >= K * N) return;
  int k = i % K; int n = i / K;
  d[i] = (bf16)s[(size_t)k * N + n];
}
// (B,T,H*HD) f32 -> (B,H,HD,T) bf16
__global__ void k_cast_vT(const float* __restrict__ s, bf16* __restrict__ d,
                          int T, int nheads) {
  int i = blockIdx.x * blockDim.x + threadIdx.x;
  int total = Bc * nheads * HDc * T;
  if (i >= total) return;
  int t = i % T;
  int r = i / T;
  int hd = r % HDc;
  int h  = (r / HDc) % nheads;
  int b  = r / (HDc * nheads);
  d[i] = (bf16)s[((size_t)b * T + t) * (nheads * HDc) + h * HDc + hd];
}
__global__ void k_zero(float* __restrict__ d, int n) {
  int i = blockIdx.x * blockDim.x + threadIdx.x;
  if (i < n) d[i] = 0.f;
}
__global__ void k_softmax4(const float* __restrict__ Lg, float* __restrict__ G4, int rows) {
  int r = blockIdx.x * blockDim.x + threadIdx.x;
  if (r >= rows) return;
  float a0 = Lg[r*4+0], a1 = Lg[r*4+1], a2 = Lg[r*4+2], a3 = Lg[r*4+3];
  float m = fmaxf(fmaxf(a0, a1), fmaxf(a2, a3));
  float e0 = __expf(a0-m), e1 = __expf(a1-m), e2 = __expf(a2-m), e3 = __expf(a3-m);
  float inv = 1.f / (e0+e1+e2+e3);
  G4[r*4+0] = e0*inv; G4[r*4+1] = e1*inv; G4[r*4+2] = e2*inv; G4[r*4+3] = e3*inv;
}
__global__ void k_moe_acc(float* __restrict__ out, const float* __restrict__ y,
                          const float* __restrict__ gate, int e, int rows, int Cd, int Ed) {
  int i = blockIdx.x * blockDim.x + threadIdx.x;
  if (i >= rows * Cd) return;
  out[i] += gate[(i / Cd) * Ed + e] * y[i];
}
__global__ void k_build_xin(const float* __restrict__ xr, const float* __restrict__ up,
                            float* __restrict__ xin, int T1, int SD, int UP, int B) {
  int i = blockIdx.x * blockDim.x + threadIdx.x;
  int W = SD + UP;
  if (i >= B * T1 * W) return;
  int c = i % W; int row = i / W; int b = row / T1;
  xin[i] = (c < SD) ? xr[row * SD + c] : up[b * UP + (c - SD)];
}
__global__ void k_slice_out(const float* __restrict__ src, float* __restrict__ dst,
                            const int* __restrict__ nn_p, int T1, int B, int Od) {
  int nn = *nn_p;
  int total = B * nn * Od;
  int i = blockIdx.x * blockDim.x + threadIdx.x;
  if (i >= total) return;
  int c = i % Od; int row = i / Od; int t = row % nn; int b = row / nn;
  dst[i] = src[((size_t)b * T1 + t) * Od + c];
}

// ---------------------------------------------------------------------------
// Host-side orchestration
// ---------------------------------------------------------------------------
struct Ws {
  float *x, *z0, *z1, *ta, *tb, *qf, *kf, *vf, *obuf, *moeout, *ybuf;
  float *h1, *h2, *glog, *gate, *xin, *outtmp;
  bf16  *bA, *bB, *bX, *qb, *kb, *vt;
};

static inline int cdiv(int a, int b) { return (a + b - 1) / b; }

static void cast_bf16(hipStream_t st, const float* s, bf16* d, int n) {
  k_cast<<<cdiv(n, 256), 256, 0, st>>>(s, d, n);
}
static void castT_bf16(hipStream_t st, const float* s, bf16* d, int K, int N) {
  k_castT<<<cdiv(K * N, 256), 256, 0, st>>>(s, d, K, N);
}
static void zero_f(hipStream_t st, float* d, int n) {
  k_zero<<<cdiv(n, 256), 256, 0, st>>>(d, n);
}
static void gemm(hipStream_t st, Ws& w, const float* A, const float* Bw, const float* bias,
                 float* Cout, const float* res, int M, int K, int N, int flags, bf16* aBuf) {
  cast_bf16(st, A, aBuf, M * K);
  castT_bf16(st, Bw, w.bB, K, N);               // bB holds B^T (N,K)
  dim3 g(cdiv(N, 64), cdiv(M, 128));
  k_gemm<<<g, 256, 0, st>>>(aBuf, w.bB, bias, Cout, res, M, K, N, flags);
}
static void mlp(hipStream_t st, Ws& w, const float* in, int Kin,
                const float* pw, const float* pb, const float* rw, const float* rb,
                const float* ow, const float* ob, int Nout, float* out) {
  gemm(st, w, in, pw, pb, w.ta, nullptr, MR, Kin, Cc, 1, w.bX);
  float* cur = w.ta; float* nxt = w.tb;
  for (int i = 0; i < MLc; ++i) {
    gemm(st, w, cur, rw + (size_t)i * Cc * Cc, rb + (size_t)i * Cc, nxt, cur, MR, Cc, Cc, 1, w.bX);
    float* t = cur; cur = nxt; nxt = t;
  }
  gemm(st, w, cur, ow, ob, out, nullptr, MR, Cc, Nout, 0, w.bX);
}
static void lnorm(hipStream_t st, const float* X, const float* g, const float* b,
                  const float* res, float* Y) {
  k_layernorm<<<cdiv(MR, 8), 256, 0, st>>>(X, g, b, res, Y, MR, Cc);
}
static void moe(hipStream_t st, Ws& w, const float* x, const float* gate,
                const float* w1, const float* b1, const float* w2, const float* b2,
                float* out) {
  zero_f(st, out, MR * Cc);
  cast_bf16(st, x, w.bX, MR * Cc);
  for (int e = 0; e < Ec; ++e) {
    castT_bf16(st, w1 + (size_t)e * Cc * Ic, w.bB, Cc, Ic);
    dim3 g1(cdiv(Ic, 64), cdiv(MR, 128));
    k_gemm<<<g1, 256, 0, st>>>(w.bX, w.bB, b1 + (size_t)e * Ic, w.h1, nullptr, MR, Cc, Ic, 1);
    cast_bf16(st, w.h1, w.bA, MR * Ic);
    castT_bf16(st, w2 + (size_t)e * Ic * Cc, w.bB, Ic, Cc);
    dim3 g2(cdiv(Cc, 64), cdiv(MR, 128));
    k_gemm<<<g2, 256, 0, st>>>(w.bA, w.bB, b2 + (size_t)e * Cc, w.ybuf, nullptr, MR, Ic, Cc, 0);
    k_moe_acc<<<cdiv(MR * Cc, 256), 256, 0, st>>>(out, w.ybuf, gate, e, MR, Cc, Ec);
  }
}

extern "C" void kernel_launch(void* const* d_in, const int* in_sizes, int n_in,
                              void* d_out, int out_size, void* d_ws, size_t ws_size,
                              hipStream_t stream) {
  (void)in_sizes; (void)n_in; (void)out_size; (void)ws_size;
  // inputs (setup_inputs order)
  const float* x_raw       = (const float*)d_in[0];
  const float* u_p         = (const float*)d_in[1];
  const float* inputs      = (const float*)d_in[2];
  const float* trunk_pre_w = (const float*)d_in[3];
  const float* trunk_pre_b = (const float*)d_in[4];
  const float* trunk_res_w = (const float*)d_in[5];
  const float* trunk_res_b = (const float*)d_in[6];
  const float* trunk_post_w= (const float*)d_in[7];
  const float* trunk_post_b= (const float*)d_in[8];
  const float* br_pre_w    = (const float*)d_in[9];
  const float* br_pre_b    = (const float*)d_in[10];
  const float* br_res_w    = (const float*)d_in[11];
  const float* br_res_b    = (const float*)d_in[12];
  const float* br_post_w   = (const float*)d_in[13];
  const float* br_post_b   = (const float*)d_in[14];
  const float* ln1_g = (const float*)d_in[15];
  const float* ln1_b = (const float*)d_in[16];
  const float* ln2_g = (const float*)d_in[17];
  const float* ln2_b = (const float*)d_in[18];
  const float* ln3_g = (const float*)d_in[19];
  const float* ln3_b = (const float*)d_in[20];
  const float* ln4_g = (const float*)d_in[21];
  const float* ln4_b = (const float*)d_in[22];
  const float* ln5_g = (const float*)d_in[23];
  const float* ln5_b = (const float*)d_in[24];
  const float* ca_q_w = (const float*)d_in[25];
  const float* ca_q_b = (const float*)d_in[26];
  const float* ca_k_w = (const float*)d_in[27];
  const float* ca_k_b = (const float*)d_in[28];
  const float* ca_v_w = (const float*)d_in[29];
  const float* ca_v_b = (const float*)d_in[30];
  const float* ca_p_w = (const float*)d_in[31];
  const float* ca_p_b = (const float*)d_in[32];
  const float* sa_q_w = (const float*)d_in[33];
  const float* sa_q_b = (const float*)d_in[34];
  const float* sa_k_w = (const float*)d_in[35];
  const float* sa_k_b = (const float*)d_in[36];
  const float* sa_v_w = (const float*)d_in[37];
  const float* sa_v_b = (const float*)d_in[38];
  const float* sa_p_w = (const float*)d_in[39];
  const float* sa_p_b = (const float*)d_in[40];
  const float* moe1_w1 = (const float*)d_in[41];
  const float* moe1_b1 = (const float*)d_in[42];
  const float* moe1_w2 = (const float*)d_in[43];
  const float* moe1_b2 = (const float*)d_in[44];
  const float* moe2_w1 = (const float*)d_in[45];
  const float* moe2_b1 = (const float*)d_in[46];
  const float* moe2_w2 = (const float*)d_in[47];
  const float* moe2_b2 = (const float*)d_in[48];
  const float* g1_w = (const float*)d_in[49];
  const float* g1_b = (const float*)d_in[50];
  const float* g2_w = (const float*)d_in[51];
  const float* g2_b = (const float*)d_in[52];
  const float* g3_w = (const float*)d_in[53];
  const float* g3_b = (const float*)d_in[54];
  const float* out_pre_w  = (const float*)d_in[55];
  const float* out_pre_b  = (const float*)d_in[56];
  const float* out_res_w  = (const float*)d_in[57];
  const float* out_res_b  = (const float*)d_in[58];
  const float* out_post_w = (const float*)d_in[59];
  const float* out_post_b = (const float*)d_in[60];
  const int*   num_nodes  = (const int*)d_in[61];

  // workspace bump allocator (256B aligned)
  char* p = (char*)d_ws;
  auto alloc = [&](size_t bytes) -> void* {
    void* r = p; p += (bytes + 255) & ~(size_t)255; return r;
  };
  Ws w;
  w.x      = (float*)alloc((size_t)MR * Cc * 4);
  w.z0     = (float*)alloc((size_t)MR * Cc * 4);
  w.z1     = (float*)alloc((size_t)MR * Cc * 4);
  w.ta     = (float*)alloc((size_t)MR * Cc * 4);
  w.tb     = (float*)alloc((size_t)MR * Cc * 4);
  w.qf     = (float*)alloc((size_t)MR * Cc * 4);
  w.kf     = (float*)alloc((size_t)MR * Cc * 4);
  w.vf     = (float*)alloc((size_t)MR * Cc * 4);
  w.obuf   = (float*)alloc((size_t)MR * Cc * 4);
  w.moeout = (float*)alloc((size_t)MR * Cc * 4);
  w.ybuf   = (float*)alloc((size_t)MR * Cc * 4);
  w.h1     = (float*)alloc((size_t)MR * Ic * 4);
  w.h2     = (float*)alloc((size_t)MR * Ic * 4);
  w.glog   = (float*)alloc((size_t)MR * Ec * 4);
  w.gate   = (float*)alloc((size_t)MR * Ec * 4);
  w.xin    = (float*)alloc((size_t)MR * 4 * 4);
  w.outtmp = (float*)alloc((size_t)MR * OUTc * 4);
  w.bA     = (bf16*)alloc((size_t)MR * Ic * 2);
  w.bB     = (bf16*)alloc((size_t)Ic * Ic * 2);
  w.bX     = (bf16*)alloc((size_t)MR * Cc * 2);
  w.qb     = (bf16*)alloc((size_t)MR * Cc * 2);
  w.kb     = (bf16*)alloc((size_t)MR * Cc * 2);
  w.vt     = (bf16*)alloc((size_t)MR * Cc * 2);   // (B,H,HD,T) transposed V

  const float scale = 0.17677669529663687f;  // 1/sqrt(HD)

  // x = trunk_mlp(concat(x_raw, bcast(u_p)))
  k_build_xin<<<cdiv(MR * 4, 256), 256, 0, stream>>>(x_raw, u_p, w.xin, T1c, SDc, 2, Bc);
  mlp(stream, w, w.xin, SDc + 2, trunk_pre_w, trunk_pre_b, trunk_res_w, trunk_res_b,
      trunk_post_w, trunk_post_b, Cc, w.x);

  // z[i] = branch_mlp_i(inputs)
  float* zbuf[NINc] = { w.z0, w.z1 };
  for (int i = 0; i < NINc; ++i) {
    mlp(stream, w, inputs, 3,
        br_pre_w + (size_t)i * 3 * Cc, br_pre_b + (size_t)i * Cc,
        br_res_w + (size_t)i * MLc * Cc * Cc, br_res_b + (size_t)i * MLc * Cc,
        br_post_w + (size_t)i * Cc * Cc, br_post_b + (size_t)i * Cc, Cc, zbuf[i]);
  }

  for (int l = 0; l < Lc; ++l) {
    // ---- expert gate from pos (== x_raw since last dim == SD) ----
    gemm(stream, w, x_raw, g1_w + (size_t)l * SDc * Ic, g1_b + (size_t)l * Ic,
         w.h1, nullptr, MR, SDc, Ic, 1, w.bX);
    gemm(stream, w, w.h1, g2_w + (size_t)l * Ic * Ic, g2_b + (size_t)l * Ic,
         w.h2, nullptr, MR, Ic, Ic, 1, w.bA);
    gemm(stream, w, w.h2, g3_w + (size_t)l * Ic * Ec, g3_b + (size_t)l * Ec,
         w.glog, nullptr, MR, Ic, Ec, 0, w.bA);
    k_softmax4<<<cdiv(MR, 256), 256, 0, stream>>>(w.glog, w.gate, MR);

    // ---- cross attention over NIN sources ----
    lnorm(stream, w.x, ln1_g + (size_t)l * Cc, ln1_b + (size_t)l * Cc, nullptr, w.ta);
    gemm(stream, w, w.ta, ca_q_w + (size_t)l * Cc * Cc, ca_q_b + (size_t)l * Cc,
         w.qf, nullptr, MR, Cc, Cc, 0, w.bX);
    cast_bf16(stream, w.qf, w.qb, MR * Cc);
    zero_f(stream, w.obuf, MR * Cc);
    for (int i = 0; i < NINc; ++i) {
      size_t wo = ((size_t)l * NINc + i);
      lnorm(stream, zbuf[i], ln2_g + wo * Cc, ln2_b + wo * Cc, nullptr, w.tb);
      gemm(stream, w, w.tb, ca_k_w + wo * Cc * Cc, ca_k_b + wo * Cc,
           w.kf, nullptr, MR, Cc, Cc, 0, w.bX);
      gemm(stream, w, w.tb, ca_v_w + wo * Cc * Cc, ca_v_b + wo * Cc,
           w.vf, nullptr, MR, Cc, Cc, 0, w.bX);
      cast_bf16(stream, w.kf, w.kb, MR * Cc);
      k_cast_vT<<<cdiv(MR * Cc, 256), 256, 0, stream>>>(w.vf, w.vt, T2c, Hc);
      dim3 ga(T1c / 128, Bc * Hc);
      k_attn<<<ga, 256, 0, stream>>>(w.qb, w.kb, w.vt, w.obuf, T1c, T2c, Cc, Hc, scale);
    }
    // x = x + o @ ca_p + b
    gemm(stream, w, w.obuf, ca_p_w + (size_t)l * Cc * Cc, ca_p_b + (size_t)l * Cc,
         w.x, w.x, MR, Cc, Cc, 0, w.bX);

    // ---- MoE 1 + LN residual ----
    moe(stream, w, w.x, w.gate,
        moe1_w1 + (size_t)l * Ec * Cc * Ic, moe1_b1 + (size_t)l * Ec * Ic,
        moe1_w2 + (size_t)l * Ec * Ic * Cc, moe1_b2 + (size_t)l * Ec * Cc, w.moeout);
    lnorm(stream, w.moeout, ln3_g + (size_t)l * Cc, ln3_b + (size_t)l * Cc, w.x, w.x);

    // ---- self attention ----
    lnorm(stream, w.x, ln4_g + (size_t)l * Cc, ln4_b + (size_t)l * Cc, nullptr, w.ta);
    gemm(stream, w, w.ta, sa_q_w + (size_t)l * Cc * Cc, sa_q_b + (size_t)l * Cc,
         w.qf, nullptr, MR, Cc, Cc, 0, w.bX);
    gemm(stream, w, w.ta, sa_k_w + (size_t)l * Cc * Cc, sa_k_b + (size_t)l * Cc,
         w.kf, nullptr, MR, Cc, Cc, 0, w.bX);
    gemm(stream, w, w.ta, sa_v_w + (size_t)l * Cc * Cc, sa_v_b + (size_t)l * Cc,
         w.vf, nullptr, MR, Cc, Cc, 0, w.bX);
    cast_bf16(stream, w.qf, w.qb, MR * Cc);
    cast_bf16(stream, w.kf, w.kb, MR * Cc);
    k_cast_vT<<<cdiv(MR * Cc, 256), 256, 0, stream>>>(w.vf, w.vt, T1c, Hc);
    zero_f(stream, w.obuf, MR * Cc);
    dim3 gs(T1c / 128, Bc * Hc);
    k_attn<<<gs, 256, 0, stream>>>(w.qb, w.kb, w.vt, w.obuf, T1c, T1c, Cc, Hc, scale);
    gemm(stream, w, w.obuf, sa_p_w + (size_t)l * Cc * Cc, sa_p_b + (size_t)l * Cc,
         w.x, w.x, MR, Cc, Cc, 0, w.bX);

    // ---- MoE 2 + LN residual ----
    moe(stream, w, w.x, w.gate,
        moe2_w1 + (size_t)l * Ec * Cc * Ic, moe2_b1 + (size_t)l * Ec * Ic,
        moe2_w2 + (size_t)l * Ec * Ic * Cc, moe2_b2 + (size_t)l * Ec * Cc, w.moeout);
    lnorm(stream, w.moeout, ln5_g + (size_t)l * Cc, ln5_b + (size_t)l * Cc, w.x, w.x);
  }

  // output MLP and num_nodes slice
  mlp(stream, w, w.x, Cc, out_pre_w, out_pre_b, out_res_w, out_res_b,
      out_post_w, out_post_b, OUTc, w.outtmp);
  k_slice_out<<<cdiv(MR * OUTc, 256), 256, 0, stream>>>(
      w.outtmp, (float*)d_out, num_nodes, T1c, Bc, OUTc);
}
